// ATDecoder_65343632441417
// MI455X (gfx1250) — matrix operand order repaired and checked by reading, not compiled, not run on hardware
//
#include <hip/hip_runtime.h>
#include <hip/hip_bf16.h>

// ---------------------------------------------------------------------------
// Types for CDNA5 WMMA (gfx1250, wave32)
// ---------------------------------------------------------------------------
typedef __attribute__((ext_vector_type(16))) __bf16 v16bf;
typedef __attribute__((ext_vector_type(8)))  __bf16 v8bf;
typedef __attribute__((ext_vector_type(8)))  float  v8f;

__device__ __forceinline__ __bf16 f2bf(float f) {
  unsigned u = __builtin_bit_cast(unsigned, f);
  unsigned r = (u + 0x7FFFu + ((u >> 16) & 1u)) >> 16;
  unsigned short h = (unsigned short)r;
  return __builtin_bit_cast(__bf16, h);
}

__device__ __forceinline__ float eluf(float x) {
  return x > 0.f ? x : expm1f(x);
}

// rsqrt(1 + 1e-5) for eval-mode BN with identity stats
#define BN_RSQ 0.9999950000374997f

// ---------------------------------------------------------------------------
// LDS helpers: 32-bit LDS byte offset of a generic pointer into shared mem,
// and the CDNA5 async global->LDS copy (ASYNCcnt-tracked, 16B per lane).
// ---------------------------------------------------------------------------
__device__ __forceinline__ unsigned lds_off_u32(const __bf16* p) {
  return (unsigned)(unsigned long long)
         (__attribute__((address_space(3))) const __bf16*)p;
}

__device__ __forceinline__ void async_copy_b128(unsigned lds_byte_off,
                                                const __bf16* gptr) {
  asm volatile("global_load_async_to_lds_b128 %0, %1, off"
               :: "v"(lds_byte_off), "v"(gptr) : "memory");
}

__device__ __forceinline__ void wait_async_all() {
  asm volatile("s_wait_asynccnt 0x0" ::: "memory");
}

// ---------------------------------------------------------------------------
// Fragment loaders (contiguous-K order, same for A and B)
// A 16x32 bf16: lane l holds row (l&15); K = khalf..khalf+15 contiguous
// with khalf = (l>>4)*16  -> two contiguous 8-half loads
// ---------------------------------------------------------------------------
__device__ __forceinline__ v16bf load_a_frag(const __bf16* __restrict__ A,
                                             int row, int lda, int kk, int lane) {
  int khalf = (lane >> 4) * 16;
  const __bf16* p = A + (size_t)row * lda + kk + khalf;
  v8bf lo = *(const v8bf*)p;
  v8bf hi = *(const v8bf*)(p + 8);
  v16bf f;
#pragma unroll
  for (int i = 0; i < 8; ++i) { f[i] = lo[i]; f[8 + i] = hi[i]; }
  return f;
}

// B 32x16 bf16 from a [64 x K] LDS tile in W layout (col-contiguous-K):
// lane l holds col (l&15); K = kb..kb+15 contiguous -> two ds_load_b128
__device__ __forceinline__ v16bf load_b_lds(const __bf16* Bs,
                                            int colL, int K, int kk, int lane) {
  int kb = kk + (lane >> 4) * 16;
  const __bf16* p = Bs + (size_t)colL * K + kb;
  v8bf lo = *(const v8bf*)p;
  v8bf hi = *(const v8bf*)(p + 8);
  v16bf f;
#pragma unroll
  for (int i = 0; i < 8; ++i) { f[i] = lo[i]; f[8 + i] = hi[i]; }
  return f;
}

// B 32x16 bf16 from K-major storage ([K,N] row-major), element (k,n)=B[k*ldb+n]
__device__ __forceinline__ v16bf load_b_fragKM(const __bf16* __restrict__ B,
                                               int ncol, int ldb, int kk, int lane) {
  int kb = kk + (lane >> 4) * 16;
  v16bf f;
#pragma unroll
  for (int i = 0; i < 16; ++i) f[i] = B[(size_t)(kb + i) * ldb + ncol];
  return f;
}

// ---------------------------------------------------------------------------
// Epilogue selector
// ---------------------------------------------------------------------------
enum {
  EPI_RELU_BF = 0,   // outB = bf( relu(acc + bias[n]) )
  EPI_ELU_BF  = 1,   // outB = bf( elu (acc + bias[n]) )
  EPI_ELU_BN  = 2,   // outB = bf( elu(acc+bias)*scale[n]*BN_RSQ + shift[n] )
  EPI_DUALMSG = 3,   // outB = bf( relu(acc+b)*rt[2r] + relu(acc2+b2)*rt[2r+1] )
  EPI_AGG_BF  = 4,   // outB = bf( acc )          (no bias)
  EPI_ADDX_F  = 5    // outF = xres + acc + bias  (only n < Nreal)
};

// ---------------------------------------------------------------------------
// Tiled WMMA GEMM: C[M,N] = epi( A[M,K] @ B^T )
//   block = 256 threads = 8 waves; wave w -> rows [mbase, mbase+16),
//   block covers 128 rows x 64 cols; grid.z batches B/C (agg GEMM).
//   W-layout path stages B (and B2) into LDS via async copies once per block.
//   Non-DUAL path uses a copy-free ping-pong pipeline (two fragment sets).
// ---------------------------------------------------------------------------
template<int EPI, bool BKMAJOR>
__global__ __launch_bounds__(256) void gemm_wmma(
    const __bf16* __restrict__ A,  const __bf16* __restrict__ B,
    const __bf16* __restrict__ A2, const __bf16* __restrict__ B2,
    const float* __restrict__ bias, const float* __restrict__ bias2,
    const float* __restrict__ rt,   const float* __restrict__ scale,
    const float* __restrict__ shift,const float* __restrict__ xres,
    __bf16* __restrict__ outB, float* __restrict__ outF,
    int M, int Nreal, int Npad, int K, int ldb, int ldc,
    long long strideBz, long long strideCz)
{
  constexpr bool DUAL = (EPI == EPI_DUALMSG);
  extern __shared__ __bf16 smem[];          // [64*K] (+ [64*K] when DUAL)

  const int lane  = threadIdx.x & 31;
  const int wave  = threadIdx.x >> 5;
  const int mbase = blockIdx.y * 128 + wave * 16;
  const int nblk  = blockIdx.x * 64;
  const int z     = blockIdx.z;

  const __bf16* Bz = B + (BKMAJOR ? (long long)z * strideBz : 0);
  const long long coff = (long long)z * strideCz;

  if (!BKMAJOR) {
    // ---- async-stage the 64-column B strip (all K) into LDS ----
    const int chunksPerCol = K >> 3;        // 8-half (16B) chunks per column
    const int total = 64 * chunksPerCol;
    for (int q = threadIdx.x; q < total; q += 256) {
      int col = q / chunksPerCol;
      int ko  = (q - col * chunksPerCol) << 3;
      int gcol = nblk + col;
      if (gcol > Npad - 1) gcol = Npad - 1;
      async_copy_b128(lds_off_u32(smem + (size_t)col * K + ko),
                      B + (size_t)gcol * ldb + ko);
      if (DUAL)
        async_copy_b128(lds_off_u32(smem + (size_t)(64 + col) * K + ko),
                        B2 + (size_t)gcol * ldb + ko);
    }
    wait_async_all();
    __syncthreads();
  }

  int mload = mbase + (lane & 15);
  if (mload >= M) mload = M - 1;   // safe clamp; stores are guarded below

  v8f c[4];  v8f c2[4];
#pragma unroll
  for (int j = 0; j < 4; ++j) { c[j] = (v8f)0.f; c2[j] = (v8f)0.f; }

  const __bf16* Bs  = smem;
  const __bf16* Bs2 = smem + (size_t)64 * K;

  // fragment fetch for K-step kk, column tile j
  auto fetchB = [&](int kk, int j) -> v16bf {
    if (BKMAJOR) {
      int nc = nblk + j * 16 + (lane & 15);
      if (nc > Npad - 1) nc = Npad - 1;
      return load_b_fragKM(Bz, nc, ldb, kk, lane);
    }
    return load_b_lds(Bs, j * 16 + (lane & 15), K, kk, lane);
  };

  if (!DUAL) {
    // ---- copy-free ping-pong pipeline over pairs of K-steps ----
    const int nsteps = K >> 5;
    v16bf aA = load_a_frag(A, mload, K, 0, lane);
    v16bf bA[4];
#pragma unroll
    for (int j = 0; j < 4; ++j) bA[j] = fetchB(0, j);

    int s = 0;
    for (; s + 2 <= nsteps; s += 2) {
      const int k1 = (s + 1) << 5;
      const int k2 = (s + 2) << 5;
      // prefetch set B for step s+1
      v16bf aB = load_a_frag(A, mload, K, k1, lane);
      v16bf bB[4];
#pragma unroll
      for (int j = 0; j < 4; ++j) bB[j] = fetchB(k1, j);
      // consume set A (step s)
#pragma unroll
      for (int j = 0; j < 4; ++j)
        c[j] = __builtin_amdgcn_wmma_f32_16x16x32_bf16(
                   false, aA, false, bA[j], (short)0, c[j], false, false);
      // prefetch set A for step s+2 (unconditional; one-past-K reads are
      // memory-safe: next row / adjacent ws region, LDS OOB reads return 0,
      // and the values are never consumed when k2 == K)
      aA = load_a_frag(A, mload, K, k2, lane);
#pragma unroll
      for (int j = 0; j < 4; ++j) bA[j] = fetchB(k2, j);
      // consume set B (step s+1)
#pragma unroll
      for (int j = 0; j < 4; ++j)
        c[j] = __builtin_amdgcn_wmma_f32_16x16x32_bf16(
                   false, aB, false, bB[j], (short)0, c[j], false, false);
    }
    if (s < nsteps) {           // odd tail: set A already holds step s
#pragma unroll
      for (int j = 0; j < 4; ++j)
        c[j] = __builtin_amdgcn_wmma_f32_16x16x32_bf16(
                   false, aA, false, bA[j], (short)0, c[j], false, false);
    }
  } else {
    // ---- dual message GEMM: two operand streams, simple schedule ----
    for (int kk = 0; kk < K; kk += 32) {
      v16bf a  = load_a_frag(A,  mload, K, kk, lane);
      v16bf a2 = load_a_frag(A2, mload, K, kk, lane);
#pragma unroll
      for (int j = 0; j < 4; ++j) {
        v16bf b  = load_b_lds(Bs,  j * 16 + (lane & 15), K, kk, lane);
        c[j] = __builtin_amdgcn_wmma_f32_16x16x32_bf16(
                   false, a, false, b, (short)0, c[j], false, false);
        v16bf b2 = load_b_lds(Bs2, j * 16 + (lane & 15), K, kk, lane);
        c2[j] = __builtin_amdgcn_wmma_f32_16x16x32_bf16(
                    false, a2, false, b2, (short)0, c2[j], false, false);
      }
    }
  }

  if (mbase >= M) return;          // M is always a multiple of 16 here

#pragma unroll
  for (int j = 0; j < 4; ++j) {
    const int ntile = nblk + j * 16;
    if (ntile >= Npad) continue;
#pragma unroll
    for (int r = 0; r < 8; ++r) {
      // C/D layout: VGPR r -> M = r (lanes 0-15) / r+8 (lanes 16-31)
      const int row = mbase + r + ((lane >> 4) << 3);
      const int col = ntile + (lane & 15);
      if (col >= Npad) continue;
      const float acc = c[j][r];
      const long long oi = coff + (long long)row * ldc + col;
      if (EPI == EPI_RELU_BF) {
        float v = (col < Nreal) ? fmaxf(acc + bias[col], 0.f) : 0.f;
        outB[oi] = f2bf(v);
      } else if (EPI == EPI_ELU_BF) {
        float v = (col < Nreal) ? eluf(acc + bias[col]) : 0.f;
        outB[oi] = f2bf(v);
      } else if (EPI == EPI_ELU_BN) {
        float v = 0.f;
        if (col < Nreal) {
          float t = eluf(acc + bias[col]);
          v = t * scale[col] * BN_RSQ + shift[col];
        }
        outB[oi] = f2bf(v);
      } else if (EPI == EPI_DUALMSG) {
        float m0 = fmaxf(acc       + bias [col], 0.f);
        float m1 = fmaxf(c2[j][r]  + bias2[col], 0.f);
        float v  = m0 * rt[2 * row] + m1 * rt[2 * row + 1];
        outB[oi] = f2bf(v);
      } else if (EPI == EPI_AGG_BF) {
        outB[oi] = f2bf(acc);
      } else { // EPI_ADDX_F
        if (col < Nreal)
          outF[oi] = xres[(long long)row * ldc + col] + acc + bias[col];
      }
    }
  }
}

// ---------------------------------------------------------------------------
// Packing / small kernels
// ---------------------------------------------------------------------------

// f32 [Nout,Kin] -> bf16 [Npad,Kpad] zero padded
__global__ void pack_w_kernel(const float* __restrict__ W, __bf16* __restrict__ out,
                              int Nout, int Kin, int Npad, int Kpad) {
  int idx = blockIdx.x * 256 + threadIdx.x;
  if (idx >= Npad * Kpad) return;
  int n = idx / Kpad, k = idx - n * Kpad;
  float v = (n < Nout && k < Kin) ? W[n * Kin + k] : 0.f;
  out[idx] = f2bf(v);
}

// rel_rec [E=2256, N=48] f32 -> rrT bf16 [48, 2272] (K zero padded)
__global__ void pack_rrT_kernel(const float* __restrict__ rel_rec,
                                __bf16* __restrict__ out) {
  int idx = blockIdx.x * 256 + threadIdx.x;
  if (idx >= 48 * 2272) return;
  int n = idx / 2272, e = idx - n * 2272;
  float v = (e < 2256) ? rel_rec[e * 48 + n] : 0.f;
  out[idx] = f2bf(v);
}

// pre_msg: row = bt*E + e.  pre[row, 0:4]=send·x, [4:8]=recv·x, [8:32]=0 (bf16)
__global__ void pre_msg_kernel(const float* __restrict__ x,
                               const float* __restrict__ rel_rec,
                               const float* __restrict__ rel_send,
                               __bf16* __restrict__ pre) {
  int row = blockIdx.x * 256 + threadIdx.x;
  if (row >= 144384) return;
  int bt = row / 2256;
  int e  = row - bt * 2256;
  const float* xb = x + (size_t)bt * 48 * 4;
  const float* rs = rel_send + (size_t)e * 48;
  const float* rr = rel_rec  + (size_t)e * 48;
  float s[4] = {0.f, 0.f, 0.f, 0.f};
  float r[4] = {0.f, 0.f, 0.f, 0.f};
  for (int n = 0; n < 48; ++n) {
    float a = rs[n], b = rr[n];
    const float* xv = xb + n * 4;
#pragma unroll
    for (int f = 0; f < 4; ++f) { s[f] += a * xv[f]; r[f] += b * xv[f]; }
  }
  __bf16* o = pre + (size_t)row * 32;
#pragma unroll
  for (int f = 0; f < 4; ++f) { o[f] = f2bf(s[f]); o[4 + f] = f2bf(r[f]); }
#pragma unroll
  for (int k = 8; k < 32; ++k) o[k] = f2bf(0.f);
}

// aug[row, 256:260] = x[row, 0:4]; aug[row, 260:288] = 0   (cols 0:256 = agg)
__global__ void aug_fill_kernel(const float* __restrict__ x,
                                __bf16* __restrict__ aug) {
  int row = blockIdx.x * 256 + threadIdx.x;
  if (row >= 3072) return;
  __bf16* o = aug + (size_t)row * 288 + 256;
#pragma unroll
  for (int f = 0; f < 4; ++f) o[f] = f2bf(x[row * 4 + f]);
#pragma unroll
  for (int k = 4; k < 32; ++k) o[k] = f2bf(0.f);
}

// ---------------------------------------------------------------------------
// Host launcher
// ---------------------------------------------------------------------------
extern "C" void kernel_launch(void* const* d_in, const int* in_sizes, int n_in,
                              void* d_out, int out_size, void* d_ws, size_t ws_size,
                              hipStream_t stream) {
  (void)in_sizes; (void)n_in; (void)out_size; (void)ws_size;

  const float* x      = (const float*)d_in[0];
  const float* relrec = (const float*)d_in[1];
  const float* relsnd = (const float*)d_in[2];
  const float* reltyp = (const float*)d_in[3];
  const float* mf1_w  = (const float*)d_in[4];
  const float* mf1_b  = (const float*)d_in[5];
  const float* mf2_w  = (const float*)d_in[6];
  const float* mf2_b  = (const float*)d_in[7];
  const float* at1_w1 = (const float*)d_in[8];
  const float* at1_b1 = (const float*)d_in[9];
  const float* at1_w2 = (const float*)d_in[10];
  const float* at1_b2 = (const float*)d_in[11];
  const float* at1_g  = (const float*)d_in[12];
  const float* at1_be = (const float*)d_in[13];
  const float* at5_w1 = (const float*)d_in[14];
  const float* at5_b1 = (const float*)d_in[15];
  const float* at5_w2 = (const float*)d_in[16];
  const float* at5_b2 = (const float*)d_in[17];
  const float* at5_g  = (const float*)d_in[18];
  const float* at5_be = (const float*)d_in[19];
  const float* o1_w   = (const float*)d_in[20];
  const float* o1_b   = (const float*)d_in[21];
  const float* o2_w   = (const float*)d_in[22];
  const float* o2_b   = (const float*)d_in[23];
  const float* o3_w   = (const float*)d_in[24];
  const float* o3_b   = (const float*)d_in[25];
  float* out = (float*)d_out;

  const int MR = 144384;         // B*T*E
  const int MN = 3072;           // B*T*N

  // ---- workspace carve (256B aligned) ----
  char*  wsb = (char*)d_ws;
  size_t off = 0;
  auto carve = [&](size_t bytes) -> void* {
    void* p = wsb + off;
    off = (off + bytes + 255) & ~(size_t)255;
    return p;
  };
  __bf16* pre    = (__bf16*)carve((size_t)MR * 32 * 2);
  __bf16* h1a    = (__bf16*)carve((size_t)MR * 256 * 2);
  __bf16* h1b    = (__bf16*)carve((size_t)MR * 256 * 2);
  __bf16* allm   = (__bf16*)carve((size_t)MR * 256 * 2);
  __bf16* hedges = (__bf16*)carve((size_t)MR * 256 * 2 + 16 * 256 * 2); // +K pad rows
  __bf16* rrT    = (__bf16*)carve((size_t)48 * 2272 * 2);
  __bf16* aug    = (__bf16*)carve((size_t)MN * 288 * 2);
  __bf16* hn1    = (__bf16*)carve((size_t)MN * 288 * 2);
  __bf16* hn2    = (__bf16*)carve((size_t)MN * 288 * 2);
  __bf16* p1     = (__bf16*)carve((size_t)MN * 256 * 2);
  __bf16* p2     = (__bf16*)carve((size_t)MN * 256 * 2);
  __bf16* mf1w0  = (__bf16*)carve(256 * 32 * 2);
  __bf16* mf1w1  = (__bf16*)carve(256 * 32 * 2);
  __bf16* mf2w0  = (__bf16*)carve(256 * 256 * 2);
  __bf16* mf2w1  = (__bf16*)carve(256 * 256 * 2);
  __bf16* a1w1p  = (__bf16*)carve(256 * 256 * 2);
  __bf16* a1w2p  = (__bf16*)carve(256 * 256 * 2);
  __bf16* a5w1p  = (__bf16*)carve(288 * 288 * 2);
  __bf16* a5w2p  = (__bf16*)carve(288 * 288 * 2);
  __bf16* o1p    = (__bf16*)carve(256 * 288 * 2);
  __bf16* o2p    = (__bf16*)carve(256 * 256 * 2);
  __bf16* o3p    = (__bf16*)carve(16  * 256 * 2);
  __bf16* hE1    = h1a;          // reuse: h1a dead after the dual-message GEMM

  // ---- weight packing ----
  auto packW = [&](const float* W, __bf16* o, int Nout, int Kin, int Np, int Kp) {
    int tot = Np * Kp;
    pack_w_kernel<<<(tot + 255) / 256, 256, 0, stream>>>(W, o, Nout, Kin, Np, Kp);
  };
  packW(mf1_w,             mf1w0, 256,   8, 256,  32);
  packW(mf1_w + 256 * 8,   mf1w1, 256,   8, 256,  32);
  packW(mf2_w,             mf2w0, 256, 256, 256, 256);
  packW(mf2_w + 256 * 256, mf2w1, 256, 256, 256, 256);
  packW(at1_w1,            a1w1p, 256, 256, 256, 256);
  packW(at1_w2,            a1w2p, 256, 256, 256, 256);
  packW(at5_w1,            a5w1p, 260, 260, 288, 288);
  packW(at5_w2,            a5w2p, 260, 260, 288, 288);
  packW(o1_w,              o1p,   256, 260, 256, 288);
  packW(o2_w,              o2p,   256, 256, 256, 256);
  packW(o3_w,              o3p,     4, 256,  16, 256);
  pack_rrT_kernel<<<(48 * 2272 + 255) / 256, 256, 0, stream>>>(relrec, rrT);

  // ---- stage 1: pre-message gather (VALU, tiny) ----
  pre_msg_kernel<<<(MR + 255) / 256, 256, 0, stream>>>(x, relrec, relsnd, pre);

  auto launch = [&](auto kern, bool dualLds, bool bkmajor,
                    int M, int Nreal, int Npad, int K, int ldb, int ldc,
                    const __bf16* A, const __bf16* B, const __bf16* A2,
                    const __bf16* B2, const float* bias, const float* bias2,
                    const float* rt, const float* sc, const float* sh,
                    const float* xr, __bf16* oB, float* oF,
                    int batches, long long sBz, long long sCz) {
    dim3 grid((Npad + 63) / 64, (M + 127) / 128, batches);
    size_t shbytes = bkmajor ? 0
                             : (size_t)(dualLds ? 2 : 1) * 64 * K * sizeof(__bf16);
    kern<<<grid, 256, shbytes, stream>>>(A, B, A2, B2, bias, bias2, rt, sc, sh,
                                         xr, oB, oF, M, Nreal, Npad, K, ldb, ldc,
                                         sBz, sCz);
  };

  // ---- stage 2/3: mf1 per edge type (relu) ----
  launch(gemm_wmma<EPI_RELU_BF, false>, false, false, MR, 256, 256, 32, 32, 256,
         pre, mf1w0, nullptr, nullptr, mf1_b, nullptr, nullptr, nullptr, nullptr,
         nullptr, h1a, nullptr, 1, 0, 0);
  launch(gemm_wmma<EPI_RELU_BF, false>, false, false, MR, 256, 256, 32, 32, 256,
         pre, mf1w1, nullptr, nullptr, mf1_b + 256, nullptr, nullptr, nullptr,
         nullptr, nullptr, h1b, nullptr, 1, 0, 0);

  // ---- stage 4: dual mf2 GEMM + fused rel_type-weighted sum ----
  launch(gemm_wmma<EPI_DUALMSG, false>, true, false, MR, 256, 256, 256, 256, 256,
         h1a, mf2w0, h1b, mf2w1, mf2_b, mf2_b + 256, reltyp, nullptr, nullptr,
         nullptr, allm, nullptr, 1, 0, 0);

  // ---- stage 5/6: at1 edge MLP (elu, elu+BN) ----
  launch(gemm_wmma<EPI_ELU_BF, false>, false, false, MR, 256, 256, 256, 256, 256,
         allm, a1w1p, nullptr, nullptr, at1_b1, nullptr, nullptr, nullptr,
         nullptr, nullptr, hE1, nullptr, 1, 0, 0);
  launch(gemm_wmma<EPI_ELU_BN, false>, false, false, MR, 256, 256, 256, 256, 256,
         hE1, a1w2p, nullptr, nullptr, at1_b2, nullptr, nullptr, at1_g, at1_be,
         nullptr, hedges, nullptr, 1, 0, 0);

  // ---- stage 7: edge -> node aggregation (batched over 64 bt slices) ----
  launch(gemm_wmma<EPI_AGG_BF, true>, false, true, 48, 256, 256, 2272, 256, 288,
         rrT, hedges, nullptr, nullptr, nullptr, nullptr, nullptr, nullptr,
         nullptr, nullptr, aug, nullptr, 64,
         (long long)2256 * 256, (long long)48 * 288);

  // ---- stage 8: append x, zero pad ----
  aug_fill_kernel<<<(MN + 255) / 256, 256, 0, stream>>>(x, aug);

  // ---- stage 9/10: at5 node MLP ----
  launch(gemm_wmma<EPI_ELU_BF, false>, false, false, MN, 260, 288, 288, 288, 288,
         aug, a5w1p, nullptr, nullptr, at5_b1, nullptr, nullptr, nullptr,
         nullptr, nullptr, hn1, nullptr, 1, 0, 0);
  launch(gemm_wmma<EPI_ELU_BN, false>, false, false, MN, 260, 288, 288, 288, 288,
         hn1, a5w2p, nullptr, nullptr, at5_b2, nullptr, nullptr, at5_g, at5_be,
         nullptr, hn2, nullptr, 1, 0, 0);

  // ---- stage 11/12: output MLP (relu x2) ----
  launch(gemm_wmma<EPI_RELU_BF, false>, false, false, MN, 256, 256, 288, 288, 256,
         hn2, o1p, nullptr, nullptr, o1_b, nullptr, nullptr, nullptr, nullptr,
         nullptr, p1, nullptr, 1, 0, 0);
  launch(gemm_wmma<EPI_RELU_BF, false>, false, false, MN, 256, 256, 256, 256, 256,
         p1, o2p, nullptr, nullptr, o2_b, nullptr, nullptr, nullptr, nullptr,
         nullptr, p2, nullptr, 1, 0, 0);

  // ---- stage 13: final projection + residual -> f32 output ----
  launch(gemm_wmma<EPI_ADDX_F, false>, false, false, MN, 4, 16, 256, 256, 4,
         p2, o3p, nullptr, nullptr, o3_b, nullptr, nullptr, nullptr, nullptr,
         x, nullptr, out, 1, 0, 0);
}
